// TimeSformerEncoder_20495583937258
// MI455X (gfx1250) — compile-verified, matrix operand order checked by reading, not run on hardware
//
#include <hip/hip_runtime.h>
#include <hip/hip_bf16.h>
#include <math.h>

// ---------------- problem constants ----------------
#define BB     4
#define FF_    16
#define CC     3
#define HH     96
#define WW     128
#define PP     8
#define HP_    12
#define WP_    16
#define NN     192              // HP*WP
#define SEQ    (FF_*NN)         // 3072
#define TOK    (BB*SEQ)         // 12288
#define DIM_   512
#define HEADS_ 8
#define DH_    64
#define INNER_ 512
#define FFH_   2048
#define QKVN   (3*INNER_)       // 1536
#define SCALE_ 0.125f           // DH^-0.5
#define PI_F   3.14159265358979323846f

typedef _Float16 f16;
typedef __attribute__((ext_vector_type(8)))  _Float16 v8h;
typedef __attribute__((ext_vector_type(16))) _Float16 v16h;
typedef __attribute__((ext_vector_type(8)))  float    v8f;

#define CAT16(lo, hi) __builtin_shufflevector((lo), (hi), 0,1,2,3,4,5,6,7,8,9,10,11,12,13,14,15)

__device__ inline v8f wmma_f16(const v16h& a, const v16h& b, const v8f& c) {
  return __builtin_amdgcn_wmma_f32_16x16x32_f16(false, a, false, b, (short)0, c, false, false);
}
__device__ inline v8f v8f_zero() {
  v8f z;
#pragma unroll
  for (int i = 0; i < 8; ++i) z[i] = 0.0f;
  return z;
}
__device__ inline v8h v8h_zero() {
  v8h z;
#pragma unroll
  for (int i = 0; i < 8; ++i) z[i] = (f16)0.0f;
  return z;
}
__device__ inline float gelu_exact(float x) {
  return 0.5f * x * (1.0f + erff(x * 0.70710678118654752f));
}

// ---------------- RoPE tables ----------------
__global__ void rope_tables_kernel(float* fsin, float* fcos, float* isin, float* icos) {
  int idx = blockIdx.x * blockDim.x + threadIdx.x;
  const int totalF = FF_ * DH_;   // 1024
  const int totalI = NN * DH_;    // 12288
  if (idx < totalF) {
    int f = idx / DH_, d = idx % DH_;
    float inv = powf(10000.0f, -(float)(d & 31) / 32.0f);
    float ang = (float)f * inv;
    fsin[idx] = sinf(ang);
    fcos[idx] = cosf(ang);
  } else if (idx < totalF + totalI) {
    int k = idx - totalF;
    int n = k / DH_, d = k % DH_;
    int half = d >> 1;            // 0..31
    int hp = n / WP_, wp = n % WP_;
    float ang;
    if (half < 16) {
      float sc = exp2f(((float)half / 15.0f) * log2f(5.0f));
      float hv = -1.0f + 2.0f * (float)hp / (float)(HP_ - 1);
      ang = hv * sc * PI_F;
    } else {
      float sc = exp2f(((float)(half - 16) / 15.0f) * log2f(5.0f));
      float wv = -1.0f + 2.0f * (float)wp / (float)(WP_ - 1);
      ang = wv * sc * PI_F;
    }
    isin[k] = sinf(ang);
    icos[k] = cosf(ang);
  }
}

// ---------------- patch rearrange (im2col) ----------------
__global__ void patch_kernel(const float* __restrict__ video, float* __restrict__ patches) {
  int idx = blockIdx.x * blockDim.x + threadIdx.x;
  const int total = TOK * (PP * PP * CC);
  if (idx >= total) return;
  int col = idx % (PP * PP * CC);
  int tok = idx / (PP * PP * CC);
  int c  = col % CC;
  int pp = col / CC;
  int p2 = pp % PP, p1 = pp / PP;
  int b = tok / SEQ, s = tok % SEQ;
  int f = s / NN,  n = s % NN;
  int hp = n / WP_, wp = n % WP_;
  int hpix = hp * PP + p1, wpix = wp * PP + p2;
  patches[idx] = video[(((size_t)(b * FF_ + f) * CC + c) * HH + hpix) * WW + wpix];
}

// ---------------- layernorm: one wave per row ----------------
__global__ void ln_kernel(const float* __restrict__ x, const float* __restrict__ g,
                          const float* __restrict__ b, float* __restrict__ y) {
  int wave = (blockIdx.x * blockDim.x + threadIdx.x) >> 5;
  int lane = threadIdx.x & 31;
  if (wave >= TOK) return;
  const float* row = x + (size_t)wave * DIM_;
  float s = 0.0f, s2 = 0.0f;
#pragma unroll
  for (int i = 0; i < DIM_ / 32; ++i) {
    float v = row[lane + i * 32];
    s += v; s2 += v * v;
  }
#pragma unroll
  for (int m = 16; m >= 1; m >>= 1) {
    s  += __shfl_xor(s, m, 32);
    s2 += __shfl_xor(s2, m, 32);
  }
  float mu  = s / (float)DIM_;
  float var = s2 / (float)DIM_ - mu * mu;
  float r   = rsqrtf(var + 1e-5f);
  float* yr = y + (size_t)wave * DIM_;
#pragma unroll
  for (int i = 0; i < DIM_ / 32; ++i) {
    int c = lane + i * 32;
    yr[c] = (row[c] - mu) * r * g[c] + b[c];
  }
}

// ---------------- RoPE + q-scale, in place on qkv ----------------
__global__ void rope_kernel(float* __restrict__ qkv, const float* __restrict__ sinT,
                            const float* __restrict__ cosT, int timeMode) {
  int idx = blockIdx.x * blockDim.x + threadIdx.x;
  const int total = TOK * (INNER_ / 2);
  if (idx >= total) return;
  int pr  = idx % (INNER_ / 2);
  int tok = idx / (INNER_ / 2);
  int s = tok % SEQ;
  int f = s / NN, n = s % NN;
  int sidx = timeMode ? f : n;
  int d0 = (2 * pr) % DH_;      // within-head dim (same table across heads)
  const float* sr = sinT + sidx * DH_;
  const float* cr = cosT + sidx * DH_;
  float sn0 = sr[d0], cs0 = cr[d0], sn1 = sr[d0 + 1], cs1 = cr[d0 + 1];
  size_t base = (size_t)tok * QKVN + 2 * pr;
  float q0 = qkv[base], q1 = qkv[base + 1];
  qkv[base]     = SCALE_ * (q0 * cs0 - q1 * sn0);
  qkv[base + 1] = SCALE_ * (q1 * cs1 + q0 * sn1);
  float k0 = qkv[base + INNER_], k1 = qkv[base + INNER_ + 1];
  qkv[base + INNER_]     = k0 * cs0 - k1 * sn0;
  qkv[base + INNER_ + 1] = k1 * cs1 + k0 * sn1;
}

// ---------------- generic WMMA GEMM: C = A@B (+bias) (+residual), GeGLU-fused A ----------------
// block tile 64x128, 8 waves, each wave 2x2 16x16 subtiles, K step 32.
// B tile is stored TRANSPOSED in LDS ([col][k]) so B fragments load as 2x ds_load_b128.
__global__ __launch_bounds__(256) void gemm_wmma_kernel(
    const float* __restrict__ A, int lda,
    const float* __restrict__ Bw, int ldb,
    float* __restrict__ Cc, int ldc,
    const float* __restrict__ bias,
    const float* __restrict__ res,
    int K, int gluOff) {
  __shared__ f16 Alds[64][40];     // pitch 80B (16B aligned)
  __shared__ f16 BldsT[128][40];   // transposed: [col][k], pitch 80B
  int tid  = threadIdx.x;
  int lane = tid & 31, wave = tid >> 5;
  int mBase = blockIdx.y * 64;
  int nBase = blockIdx.x * 128;
  int wm = wave >> 2, wn = wave & 3;     // 2 x 4 waves
  int lr = lane & 15;
  int msel = lane >> 4;
  int ksel = msel * 8;
  v8f acc[2][2];
#pragma unroll
  for (int i = 0; i < 2; ++i)
#pragma unroll
    for (int j = 0; j < 2; ++j) acc[i][j] = v8f_zero();

  // A staging coords: 8 consecutive k per thread
  const int aRow = (tid * 8) >> 5;
  const int aCol = (tid * 8) & 31;
  // B staging coords: one column segment of 16 k per thread (coalesced across wave)
  const int bCol   = tid & 127;
  const int bKBase = (tid >> 7) * 16;

  for (int k0 = 0; k0 < K; k0 += 32) {
    // ---- stage A tile 64x32 (optionally a*gelu(g) fused), vector LDS store ----
    {
      const float* ap_ = &A[(size_t)(mBase + aRow) * lda + k0 + aCol];
      v8h pk;
#pragma unroll
      for (int t = 0; t < 8; ++t) {
        float v = ap_[t];
        if (gluOff) v *= gelu_exact(ap_[t + gluOff]);
        pk[t] = (f16)v;
      }
      *(v8h*)&Alds[aRow][aCol] = pk;
    }
    // ---- stage B tile 32x128 transposed ----
    {
      v8h p0, p1;
#pragma unroll
      for (int t = 0; t < 8; ++t)
        p0[t] = (f16)Bw[(size_t)(k0 + bKBase + t) * ldb + nBase + bCol];
#pragma unroll
      for (int t = 0; t < 8; ++t)
        p1[t] = (f16)Bw[(size_t)(k0 + bKBase + 8 + t) * ldb + nBase + bCol];
      *(v8h*)&BldsT[bCol][bKBase]     = p0;
      *(v8h*)&BldsT[bCol][bKBase + 8] = p1;
    }
    __syncthreads();

    v16h af[2], bf[2];
#pragma unroll
    for (int sm = 0; sm < 2; ++sm) {
      int mrow = wm * 32 + sm * 16 + lr;
      v8h lo = *(const v8h*)&Alds[mrow][ksel];
      v8h hi = *(const v8h*)&Alds[mrow][16 + ksel];
      af[sm] = CAT16(lo, hi);
    }
#pragma unroll
    for (int sn = 0; sn < 2; ++sn) {
      int ncol = wn * 32 + sn * 16 + lr;
      v8h lo = *(const v8h*)&BldsT[ncol][ksel];
      v8h hi = *(const v8h*)&BldsT[ncol][16 + ksel];
      bf[sn] = CAT16(lo, hi);
    }
#pragma unroll
    for (int sm = 0; sm < 2; ++sm)
#pragma unroll
      for (int sn = 0; sn < 2; ++sn)
        acc[sm][sn] = wmma_f16(af[sm], bf[sn], acc[sm][sn]);
    __syncthreads();
  }

#pragma unroll
  for (int sm = 0; sm < 2; ++sm)
#pragma unroll
    for (int sn = 0; sn < 2; ++sn)
#pragma unroll
      for (int r = 0; r < 8; ++r) {
        int row = mBase + wm * 32 + sm * 16 + msel * 8 + r;
        int col = nBase + wn * 32 + sn * 16 + lr;
        float v = acc[sm][sn][r];
        if (bias) v += bias[col];
        if (res)  v += res[(size_t)row * ldc + col];
        Cc[(size_t)row * ldc + col] = v;
      }
}

// ---------------- time attention: one wave per (b,h,n), seq=16 over frames ----------------
__global__ __launch_bounds__(128) void time_attn_kernel(const float* __restrict__ qkv,
                                                        float* __restrict__ attn) {
  __shared__ f16 plds[4][16][16];   // pitch 32B
  int lane = threadIdx.x & 31;
  int wave = threadIdx.x >> 5;
  int g   = blockIdx.x * 4 + wave;     // 6144 groups total
  int b   = g / (HEADS_ * NN);
  int rem = g % (HEADS_ * NN);
  int h = rem / NN, n = rem % NN;
  int lr = lane & 15, msel = lane >> 4, ksel = msel * 8;
  const size_t rs = (size_t)NN * QKVN;   // stride between frames for fixed n
  const float* qbase = qkv + (size_t)(b * SEQ + n) * QKVN + h * DH_;
  const float* kbase = qbase + INNER_;
  const float* vbase = qbase + 2 * INNER_;

  // sim(16x16) = q(16x64) @ k^T, two K-steps
  v8f c = v8f_zero();
#pragma unroll
  for (int kk = 0; kk < 2; ++kk) {
    int koff = kk * 32;
    const float* qrow = qbase + (size_t)lr * rs;  // A row m = frame lr
    const float* krow = kbase + (size_t)lr * rs;  // B col j = frame lr
    v8h qlo, qhi, klo, khi;
#pragma unroll
    for (int i = 0; i < 8; ++i) {
      qlo[i] = (f16)qrow[koff + ksel + i];
      qhi[i] = (f16)qrow[koff + 16 + ksel + i];
      klo[i] = (f16)krow[koff + ksel + i];
      khi[i] = (f16)krow[koff + 16 + ksel + i];
    }
    v16h a  = CAT16(qlo, qhi);
    v16h bb = CAT16(klo, khi);
    c = wmma_f16(a, bb, c);
  }

  // row softmax over 16 cols (each C row lives across a 16-lane half-group)
  float p[8];
#pragma unroll
  for (int r = 0; r < 8; ++r) {
    float v = c[r];
    float mx = v;
#pragma unroll
    for (int m = 8; m >= 1; m >>= 1) mx = fmaxf(mx, __shfl_xor(mx, m, 16));
    float e = __expf(v - mx);
    float s = e;
#pragma unroll
    for (int m = 8; m >= 1; m >>= 1) s += __shfl_xor(s, m, 16);
    p[r] = e / s;
  }
#pragma unroll
  for (int r = 0; r < 8; ++r) plds[wave][msel * 8 + r][lr] = (f16)p[r];

  // A fragment of P (16x16 padded to 16x32 K)
  v16h ap;
  {
    v8h lo = *(const v8h*)&plds[wave][lr][ksel];
    ap = CAT16(lo, v8h_zero());
  }
  // o = P @ V, 4 d-tiles
#pragma unroll
  for (int dt = 0; dt < 4; ++dt) {
    int ncol = dt * 16 + lr;
    v8h blo;
#pragma unroll
    for (int i = 0; i < 8; ++i)
      blo[i] = (f16)vbase[(size_t)(ksel + i) * rs + ncol];
    v16h bv = CAT16(blo, v8h_zero());
    v8f o = wmma_f16(ap, bv, v8f_zero());
#pragma unroll
    for (int r = 0; r < 8; ++r) {
      int fo = msel * 8 + r;
      attn[(size_t)(b * SEQ + fo * NN + n) * INNER_ + h * DH_ + ncol] = o[r];
    }
  }
}

// ---------------- space attention (flash): block = (b,h,f,qtile), 4 waves x 16 q-rows ----------------
// V chunk stored TRANSPOSED in LDS so P@V fragments load as 2x ds_load_b128.
__global__ __launch_bounds__(128) void space_attn_kernel(const float* __restrict__ qkv,
                                                         float* __restrict__ attn) {
  __shared__ f16 Klds[32][72];     // row-major [k][d], pitch 144B
  __shared__ f16 VldsT[64][40];    // transposed [d][k], pitch 80B
  __shared__ f16 Plds[4][16][40];  // per-wave 16x32, pitch 80B
  int tid = threadIdx.x;
  int lane = tid & 31, wave = tid >> 5;
  int g  = blockIdx.x / 3;
  int qt = blockIdx.x % 3;
  int b   = g / (HEADS_ * FF_);
  int rem = g % (HEADS_ * FF_);
  int h = rem / FF_, f = rem % FF_;
  int lr = lane & 15, msel = lane >> 4, ksel = msel * 8;
  size_t tokBase = (size_t)(b * SEQ + f * NN);
  const float* qptr = qkv + tokBase * QKVN + h * DH_;
  const float* kptr = qptr + INNER_;
  const float* vptr = qptr + 2 * INNER_;
  int qBase = qt * 64 + wave * 16;

  // preload q A-fragments (16x64 -> two K-steps)
  v16h aq[2];
  {
    const float* qrow = qptr + (size_t)(qBase + lr) * QKVN;
#pragma unroll
    for (int kk = 0; kk < 2; ++kk) {
      int koff = kk * 32;
      v8h lo, hi;
#pragma unroll
      for (int i = 0; i < 8; ++i) {
        lo[i] = (f16)qrow[koff + ksel + i];
        hi[i] = (f16)qrow[koff + 16 + ksel + i];
      }
      aq[kk] = CAT16(lo, hi);
    }
  }

  v8f ov[4];
#pragma unroll
  for (int dt = 0; dt < 4; ++dt) ov[dt] = v8f_zero();
  float mrow[8], lrow[8];
#pragma unroll
  for (int r = 0; r < 8; ++r) { mrow[r] = -1e30f; lrow[r] = 0.0f; }

  // staging coords
  const int kRow = (tid * 16) >> 6;          // K chunk: 16 consecutive d per thread
  const int kCol = (tid * 16) & 63;
  const int vCol   = tid & 63;               // V chunk: column segment per thread
  const int vKBase = (tid >> 6) * 16;

  for (int j0 = 0; j0 < NN; j0 += 32) {
    // ---- stage K (row-major) and V (transposed) chunks: 32 kv rows x 64 dims ----
    {
      size_t off = (size_t)(j0 + kRow) * QKVN + kCol;
      v8h a0, a1;
#pragma unroll
      for (int t = 0; t < 8; ++t) { a0[t] = (f16)kptr[off + t]; a1[t] = (f16)kptr[off + 8 + t]; }
      *(v8h*)&Klds[kRow][kCol]     = a0;
      *(v8h*)&Klds[kRow][kCol + 8] = a1;
      v8h p0, p1;
#pragma unroll
      for (int t = 0; t < 8; ++t)
        p0[t] = (f16)vptr[(size_t)(j0 + vKBase + t) * QKVN + vCol];
#pragma unroll
      for (int t = 0; t < 8; ++t)
        p1[t] = (f16)vptr[(size_t)(j0 + vKBase + 8 + t) * QKVN + vCol];
      *(v8h*)&VldsT[vCol][vKBase]     = p0;
      *(v8h*)&VldsT[vCol][vKBase + 8] = p1;
    }
    __syncthreads();

    // S = q @ K^T for 32 kv cols -> two 16x16 C tiles
    v8f s[2];
#pragma unroll
    for (int half = 0; half < 2; ++half) {
      s[half] = v8f_zero();
      int krow = half * 16 + lr;     // B col j = local kv row
#pragma unroll
      for (int kk = 0; kk < 2; ++kk) {
        int koff = kk * 32;
        v8h lo = *(const v8h*)&Klds[krow][koff + ksel];
        v8h hi = *(const v8h*)&Klds[krow][koff + 16 + ksel];
        v16h bk = CAT16(lo, hi);
        s[half] = wmma_f16(aq[kk], bk, s[half]);
      }
    }

    // online softmax per row
    float sc[8];
#pragma unroll
    for (int r = 0; r < 8; ++r) {
      float v0 = s[0][r], v1 = s[1][r];
      float mx = fmaxf(v0, v1);
#pragma unroll
      for (int m = 8; m >= 1; m >>= 1) mx = fmaxf(mx, __shfl_xor(mx, m, 16));
      float mnew = fmaxf(mrow[r], mx);
      sc[r] = __expf(mrow[r] - mnew);
      float p0 = __expf(v0 - mnew);
      float p1 = __expf(v1 - mnew);
      float ss = p0 + p1;
#pragma unroll
      for (int m = 8; m >= 1; m >>= 1) ss += __shfl_xor(ss, m, 16);
      lrow[r] = lrow[r] * sc[r] + ss;
      mrow[r] = mnew;
      Plds[wave][msel * 8 + r][lr]      = (f16)p0;
      Plds[wave][msel * 8 + r][16 + lr] = (f16)p1;
    }
#pragma unroll
    for (int dt = 0; dt < 4; ++dt)
#pragma unroll
      for (int r = 0; r < 8; ++r) ov[dt][r] *= sc[r];

    // A fragment from P (16x32), then O += P @ V
    v16h ap;
    {
      v8h lo = *(const v8h*)&Plds[wave][lr][ksel];
      v8h hi = *(const v8h*)&Plds[wave][lr][16 + ksel];
      ap = CAT16(lo, hi);
    }
#pragma unroll
    for (int dt = 0; dt < 4; ++dt) {
      int ncol = dt * 16 + lr;
      v8h lo = *(const v8h*)&VldsT[ncol][ksel];
      v8h hi = *(const v8h*)&VldsT[ncol][16 + ksel];
      v16h bv = CAT16(lo, hi);
      ov[dt] = wmma_f16(ap, bv, ov[dt]);
    }
    __syncthreads();
  }

  // normalize + write
#pragma unroll
  for (int dt = 0; dt < 4; ++dt) {
    int ncol = dt * 16 + lr;
#pragma unroll
    for (int r = 0; r < 8; ++r) {
      int row = qBase + msel * 8 + r;
      attn[(tokBase + row) * INNER_ + h * DH_ + ncol] = ov[dt][r] / lrow[r];
    }
  }
}

// ---------------- host orchestration ----------------
extern "C" void kernel_launch(void* const* d_in, const int* in_sizes, int n_in,
                              void* d_out, int out_size, void* d_ws, size_t ws_size,
                              hipStream_t stream) {
  (void)in_sizes; (void)n_in; (void)out_size; (void)ws_size;
  const float* video   = (const float*)d_in[0];
  const float* patch_w = (const float*)d_in[1];
  const float* patch_b = (const float*)d_in[2];
  const float* ln_t_g  = (const float*)d_in[3];
  const float* ln_t_b  = (const float*)d_in[4];
  const float* ln_s_g  = (const float*)d_in[5];
  const float* ln_s_b  = (const float*)d_in[6];
  const float* ln_f_g  = (const float*)d_in[7];
  const float* ln_f_b  = (const float*)d_in[8];
  const float* qkv_t_w = (const float*)d_in[9];
  const float* out_t_w = (const float*)d_in[10];
  const float* out_t_b = (const float*)d_in[11];
  const float* qkv_s_w = (const float*)d_in[12];
  const float* out_s_w = (const float*)d_in[13];
  const float* out_s_b = (const float*)d_in[14];
  const float* ff_w1   = (const float*)d_in[15];
  const float* ff_b1   = (const float*)d_in[16];
  const float* ff_w2   = (const float*)d_in[17];
  const float* ff_b2   = (const float*)d_in[18];

  float* x = (float*)d_out;                       // (TOK, DIM) residual stream
  char* ws = (char*)d_ws;
  float* y    = (float*)ws; ws += (size_t)TOK * DIM_  * sizeof(float);
  float* big  = (float*)ws; ws += (size_t)TOK * 4096  * sizeof(float);  // phase-aliased
  float* fsin = (float*)ws; ws += (size_t)FF_ * DH_ * sizeof(float);
  float* fcos = (float*)ws; ws += (size_t)FF_ * DH_ * sizeof(float);
  float* isin = (float*)ws; ws += (size_t)NN  * DH_ * sizeof(float);
  float* icos = (float*)ws; ws += (size_t)NN  * DH_ * sizeof(float);

  float* patches = big;                           // TOK x 192   (pre-layer phase)
  float* qkvB    = big;                           // TOK x 1536  (attention phase)
  float* attnB   = big + (size_t)TOK * QKVN;      // TOK x 512   (attention phase)
  float* h1      = big;                           // TOK x 4096  (FFN phase)

  auto gemm = [&](const float* A, int lda, const float* Bw, int ldb, float* Cc, int ldc,
                  const float* bias, const float* res, int M, int Nn, int K, int gluOff) {
    dim3 grid(Nn / 128, M / 64);
    gemm_wmma_kernel<<<grid, 256, 0, stream>>>(A, lda, Bw, ldb, Cc, ldc, bias, res, K, gluOff);
  };

  // RoPE tables + patchify + patch embedding
  rope_tables_kernel<<<(FF_ * DH_ + NN * DH_ + 255) / 256, 256, 0, stream>>>(fsin, fcos, isin, icos);
  {
    int total = TOK * PP * PP * CC;
    patch_kernel<<<(total + 255) / 256, 256, 0, stream>>>(video, patches);
  }
  gemm(patches, PP * PP * CC, patch_w, DIM_, x, DIM_, patch_b, nullptr, TOK, DIM_, PP * PP * CC, 0);

  const int lnBlocks   = TOK / 8;                  // 8 waves/block, 1 row/wave
  const int ropeBlocks = (TOK * (INNER_ / 2)) / 256;

  for (int l = 0; l < 4; ++l) {
    // ---- time attention ----
    ln_kernel<<<lnBlocks, 256, 0, stream>>>(x, ln_t_g + l * DIM_, ln_t_b + l * DIM_, y);
    gemm(y, DIM_, qkv_t_w + (size_t)l * DIM_ * QKVN, QKVN, qkvB, QKVN,
         nullptr, nullptr, TOK, QKVN, DIM_, 0);
    rope_kernel<<<ropeBlocks, 256, 0, stream>>>(qkvB, fsin, fcos, 1);
    time_attn_kernel<<<(BB * HEADS_ * NN) / 4, 128, 0, stream>>>(qkvB, attnB);
    gemm(attnB, INNER_, out_t_w + (size_t)l * INNER_ * DIM_, DIM_, x, DIM_,
         out_t_b + l * DIM_, x, TOK, DIM_, INNER_, 0);

    // ---- space attention ----
    ln_kernel<<<lnBlocks, 256, 0, stream>>>(x, ln_s_g + l * DIM_, ln_s_b + l * DIM_, y);
    gemm(y, DIM_, qkv_s_w + (size_t)l * DIM_ * QKVN, QKVN, qkvB, QKVN,
         nullptr, nullptr, TOK, QKVN, DIM_, 0);
    rope_kernel<<<ropeBlocks, 256, 0, stream>>>(qkvB, isin, icos, 0);
    space_attn_kernel<<<BB * HEADS_ * FF_ * 3, 128, 0, stream>>>(qkvB, attnB);
    gemm(attnB, INNER_, out_s_w + (size_t)l * INNER_ * DIM_, DIM_, x, DIM_,
         out_s_b + l * DIM_, x, TOK, DIM_, INNER_, 0);

    // ---- GeGLU FFN ----
    ln_kernel<<<lnBlocks, 256, 0, stream>>>(x, ln_f_g + l * DIM_, ln_f_b + l * DIM_, y);
    gemm(y, DIM_, ff_w1 + (size_t)l * DIM_ * 2 * FFH_, 2 * FFH_, h1, 2 * FFH_,
         ff_b1 + (size_t)l * 2 * FFH_, nullptr, TOK, 2 * FFH_, DIM_, 0);
    // second GEMM reads a = h1[:, :2048] with a*gelu(h1[:, 2048:]) fused into A staging
    gemm(h1, 2 * FFH_, ff_w2 + (size_t)l * FFH_ * DIM_, DIM_, x, DIM_,
         ff_b2 + l * DIM_, x, TOK, DIM_, FFH_, FFH_);
  }
}
